// GGNN_13486197309968
// MI455X (gfx1250) — matrix operand order repaired and checked by reading, not compiled, not run on hardware
//
#include <hip/hip_runtime.h>
#include <hip/hip_bf16.h>
#include <math.h>

typedef __attribute__((ext_vector_type(16))) _Float16 v16h;
typedef __attribute__((ext_vector_type(8)))  float    v8f;

#define D 64
#define NPG 200           // nodes per graph
#define ROWS 208          // padded to 13 * 16
#define NET 6             // edge types
#define NSTEPS 5
#define NITERS 6

// ---------- WMMA fragment helpers (wave32, this code's contiguous k = 16h + s per lane half; A and B match) ----------

// A (16x32 f16): lane<16 -> row M=lane, K in {0..15}; lane>=16 -> row M=lane-16, K+=16
__device__ inline v16h load_A_lds(const float* __restrict__ H, int ld, int m0, int kbase) {
  int lane = threadIdx.x & 31;
  const float* rowp = H + (m0 + (lane & 15)) * ld;
  int kh = (lane >> 4) ? 16 : 0;
  v16h a;
#pragma unroll
  for (int v = 0; v < 8; ++v) {
    int k0 = kbase + 2 * v + kh;
    a[2 * v]     = (_Float16)rowp[k0];
    a[2 * v + 1] = (_Float16)rowp[k0 + 1];
  }
  return a;
}

// B (32x16 f16) from row-major B[k][n] (K rows): lane<16 -> col N=lane, K=2v(+1); lane>=16 -> K+=16
__device__ inline v16h load_B_rm(const float* __restrict__ Bp, int ld, int n0, int kbase) {
  int lane = threadIdx.x & 31;
  int col = n0 + (lane & 15);
  int kh = (lane >> 4) ? 16 : 0;
  v16h b;
#pragma unroll
  for (int v = 0; v < 8; ++v) {
    int k0 = kbase + kh + 2 * v;
    b[2 * v]     = (_Float16)Bp[k0 * ld + col];
    b[2 * v + 1] = (_Float16)Bp[(k0 + 1) * ld + col];
  }
  return b;
}

// B = W^T where W is [Ncols][K] row-major: B[k][n] = W[(n0+n)*ld + k]
__device__ inline v16h load_B_tr(const float* __restrict__ Wp, int ld, int n0, int kbase) {
  int lane = threadIdx.x & 31;
  const float* colp = Wp + (n0 + (lane & 15)) * ld;
  int kh = (lane >> 4) ? 16 : 0;
  v16h b;
#pragma unroll
  for (int v = 0; v < 8; ++v) {
    int k0 = kbase + kh + 2 * v;
    b[2 * v]     = (_Float16)colp[k0];
    b[2 * v + 1] = (_Float16)colp[k0 + 1];
  }
  return b;
}

__device__ inline v8f wmma16(v16h a, v16h b, v8f c) {
  return __builtin_amdgcn_wmma_f32_16x16x32_f16(false, a, false, b, (short)0, c, false, false);
}

__device__ inline float sigmoidf(float x) { return 1.0f / (1.0f + __expf(-x)); }

// =====================================================================================
// Phase A: GGNN propagation. One workgroup per graph, all 5 steps resident in LDS.
// Dynamic LDS layout:
//   [0 .. 53248)        H0  : float [208][64]      (ping)
//   [53248 .. 106496)   H1  : float [208][64]      (pong / accumulator)
//   [106496 .. 266240)  S   : time-shared scratch:
//         phase 1-2: _Float16 Wh[6][208][64]  (159744 B)
//         phase 3-4: float    gi[208][192]    (159744 B)
// =====================================================================================
__global__ __launch_bounds__(256)
void ggnn_propagate(const float* __restrict__ feats,
                    const float* __restrict__ W_edge,   // [6][64][64] (d-major rows -> B row-major)
                    const float* __restrict__ b_edge,   // [6][64]
                    const float* __restrict__ W_ih,     // [192][64]
                    const float* __restrict__ W_hh,     // [192][64]
                    const float* __restrict__ b_ih,     // [192]
                    const float* __restrict__ b_hh,     // [192]
                    const int*   __restrict__ src,
                    const int*   __restrict__ dst,
                    const int*   __restrict__ etypes,
                    int epg,                             // edges per graph (contiguous)
                    float* __restrict__ feat_out)
{
  extern __shared__ char smem[];
  float*    H0  = (float*)smem;
  float*    H1  = (float*)(smem + ROWS * D * 4);
  char*     S   = smem + 2 * ROWS * D * 4;
  _Float16* WhS = (_Float16*)S;       // [6][208][64]
  float*    GI  = (float*)S;          // [208][192]

  const int g = blockIdx.x;
  const int tid = threadIdx.x;
  const int wave = tid >> 5;
  const int lane = tid & 31;
  const int nodebase = g * NPG;
  const int edgebase = g * epg;

  // Load node features (pad rows -> 0)
  for (int i = tid; i < ROWS * D; i += 256) {
    int r = i >> 6;
    H0[i] = (r < NPG) ? feats[(nodebase + r) * D + (i & 63)] : 0.0f;
  }
  __syncthreads();

  float* Hc = H0;   // current h
  float* Ha = H1;   // accumulator / h_new

  for (int step = 0; step < NSTEPS; ++step) {
    // ---- (1) Wh[t] = h @ W_edge[t] + b_edge[t]  ->  f16 in S ----
    for (int job = wave; job < NET * 13 * 4; job += 8) {
      int t = job / 52, rem = job % 52;
      int m0 = (rem >> 2) * 16, n0 = (rem & 3) * 16;
      v8f c = {};
#pragma unroll
      for (int kk = 0; kk < 2; ++kk) {
        v16h a = load_A_lds(Hc, D, m0, kk * 32);
        v16h b = load_B_rm(W_edge + t * D * D, D, n0, kk * 32);
        c = wmma16(a, b, c);
      }
      int colL = lane & 15, rh = (lane >> 4) * 8;
      float bias = b_edge[t * D + n0 + colL];
#pragma unroll
      for (int v = 0; v < 8; ++v) {
        int row = m0 + v + rh;
        WhS[((t * ROWS + row) * D) + n0 + colL] = (_Float16)(c[v] + bias);
      }
    }
    // zero accumulator
    for (int i = tid; i < ROWS * D; i += 256) Ha[i] = 0.0f;
    __syncthreads();

    // ---- (2) edge scatter: a[dst] += Wh[etype][src]  (ds_add_f32 atomics) ----
    {
      int c0 = lane * 2;   // 2 columns per lane (64 cols / 32 lanes)
      for (int e = wave; e < epg; e += 8) {
        int eg = edgebase + e;
        int ls = src[eg] - nodebase;
        int ld_ = dst[eg] - nodebase;
        int t = etypes[eg];
        const _Float16* wp = WhS + ((t * ROWS + ls) * D);
        atomicAdd(&Ha[ld_ * D + c0],     (float)wp[c0]);
        atomicAdd(&Ha[ld_ * D + c0 + 1], (float)wp[c0 + 1]);
      }
    }
    __syncthreads();

    // ---- (3) gi = a @ W_ih^T + b_ih  -> f32 in S (overwrites Wh) ----
    for (int job = wave; job < 13 * 12; job += 8) {
      int m0 = (job / 12) * 16, n0 = (job % 12) * 16;
      v8f c = {};
#pragma unroll
      for (int kk = 0; kk < 2; ++kk) {
        v16h a = load_A_lds(Ha, D, m0, kk * 32);
        v16h b = load_B_tr(W_ih, D, n0, kk * 32);
        c = wmma16(a, b, c);
      }
      int colL = lane & 15, rh = (lane >> 4) * 8;
      float bias = b_ih[n0 + colL];
#pragma unroll
      for (int v = 0; v < 8; ++v) {
        int row = m0 + v + rh;
        GI[row * 192 + n0 + colL] = c[v] + bias;
      }
    }
    __syncthreads();

    // ---- (4) gh tiles (r,z,n together) + fused GRU gates -> h_new into Ha ----
    for (int job = wave; job < 13 * 4; job += 8) {
      int m0 = (job >> 2) * 16, j0 = (job & 3) * 16;
      v8f cr = {}, cz = {}, cn = {};
#pragma unroll
      for (int kk = 0; kk < 2; ++kk) {
        v16h a  = load_A_lds(Hc, D, m0, kk * 32);
        v16h br = load_B_tr(W_hh, D, j0, kk * 32);
        v16h bz = load_B_tr(W_hh, D, 64 + j0, kk * 32);
        v16h bn = load_B_tr(W_hh, D, 128 + j0, kk * 32);
        cr = wmma16(a, br, cr);
        cz = wmma16(a, bz, cz);
        cn = wmma16(a, bn, cn);
      }
      int colL = lane & 15, rh = (lane >> 4) * 8;
      int col = j0 + colL;
      float bhr = b_hh[col], bhz = b_hh[64 + col], bhn = b_hh[128 + col];
#pragma unroll
      for (int v = 0; v < 8; ++v) {
        int row = m0 + v + rh;
        float gir = GI[row * 192 + col];
        float giz = GI[row * 192 + 64 + col];
        float gin = GI[row * 192 + 128 + col];
        float r = sigmoidf(gir + (cr[v] + bhr));
        float z = sigmoidf(giz + (cz[v] + bhz));
        float n = tanhf(gin + r * (cn[v] + bhn));
        float hold = Hc[row * D + col];
        float hn = (1.0f - z) * n + z * hold;
        Ha[row * D + col] = (row < NPG) ? hn : 0.0f;
      }
    }
    __syncthreads();
    float* tmp = Hc; Hc = Ha; Ha = tmp;
  }

  for (int i = tid; i < NPG * D; i += 256)
    feat_out[nodebase * D + i] = Hc[i];
}

// =====================================================================================
// Phase B: Set2Set readout — fully per-graph independent; one WG per graph.
// =====================================================================================
__global__ __launch_bounds__(256)
void set2set_readout(const float* __restrict__ feat,   // [N][64] from phase A
                     const float* __restrict__ Wi0, const float* __restrict__ Wh0,
                     const float* __restrict__ bi0, const float* __restrict__ bh0,
                     const float* __restrict__ Wi1, const float* __restrict__ Wh1,
                     const float* __restrict__ bi1, const float* __restrict__ bh1,
                     const float* __restrict__ Wi2, const float* __restrict__ Wh2,
                     const float* __restrict__ bi2, const float* __restrict__ bh2,
                     const float* __restrict__ Wp, const float* __restrict__ bp,
                     float* __restrict__ out)           // [G][3]
{
  __shared__ float F[NPG * D];        // 51200 B
  __shared__ float hs[3][D], csv[3][D];
  __shared__ float qstar[2 * D];
  __shared__ float gates[4 * D];
  __shared__ float xbuf[D];
  __shared__ float ebuf[NPG];
  __shared__ float red[2];

  const int g = blockIdx.x;
  const int tid = threadIdx.x;
  const int nodebase = g * NPG;

  for (int i = tid; i < NPG * D; i += 256) F[i] = feat[nodebase * D + i];
  for (int i = tid; i < 3 * D; i += 256) { hs[i / D][i % D] = 0.0f; csv[i / D][i % D] = 0.0f; }
  for (int i = tid; i < 2 * D; i += 256) qstar[i] = 0.0f;
  __syncthreads();

  const float* WiL[3] = {Wi0, Wi1, Wi2};
  const float* WhL[3] = {Wh0, Wh1, Wh2};
  const float* biL[3] = {bi0, bi1, bi2};
  const float* bhL[3] = {bh0, bh1, bh2};

  for (int it = 0; it < NITERS; ++it) {
    // --- 3-layer LSTM (batch row = this graph) ---
    for (int l = 0; l < 3; ++l) {
      const float* x = (l == 0) ? qstar : xbuf;
      int xdim = (l == 0) ? 2 * D : D;
      // 256 gates, one per thread
      {
        int j = tid;
        float acc = biL[l][j] + bhL[l][j];
        const float* wi = WiL[l] + j * xdim;
        const float* wh = WhL[l] + j * D;
        for (int k = 0; k < xdim; ++k) acc += x[k] * wi[k];
        for (int k = 0; k < D; ++k)    acc += hs[l][k] * wh[k];
        gates[j] = acc;
      }
      __syncthreads();
      if (tid < D) {
        int j = tid;
        float c = sigmoidf(gates[D + j]) * csv[l][j] + sigmoidf(gates[j]) * tanhf(gates[2 * D + j]);
        csv[l][j] = c;
        float nx = sigmoidf(gates[3 * D + j]) * tanhf(c);
        hs[l][j] = nx;
        xbuf[j] = nx;
      }
      __syncthreads();
    }
    // --- attention: e[n] = <feat[n], q>, segment softmax, readout ---
    if (tid < NPG) {
      const float* fr = F + tid * D;
      float acc = 0.0f;
      for (int k = 0; k < D; ++k) acc += fr[k] * xbuf[k];
      ebuf[tid] = acc;
    }
    __syncthreads();
    if (tid < 32) {
      float m = -1e30f;
      for (int i = tid; i < NPG; i += 32) m = fmaxf(m, ebuf[i]);
#pragma unroll
      for (int off = 16; off > 0; off >>= 1) m = fmaxf(m, __shfl_xor(m, off, 32));
      if (tid == 0) red[0] = m;
    }
    __syncthreads();
    if (tid < NPG) ebuf[tid] = __expf(ebuf[tid] - red[0]);
    __syncthreads();
    if (tid < 32) {
      float s = 0.0f;
      for (int i = tid; i < NPG; i += 32) s += ebuf[i];
#pragma unroll
      for (int off = 16; off > 0; off >>= 1) s += __shfl_xor(s, off, 32);
      if (tid == 0) red[1] = s;
    }
    __syncthreads();
    if (tid < D) {
      float s = 0.0f;
      for (int n = 0; n < NPG; ++n) s += F[n * D + tid] * ebuf[n];
      qstar[D + tid] = s / red[1];
      qstar[tid] = xbuf[tid];
    }
    __syncthreads();
  }

  if (tid < 3) {
    float acc = bp[tid];
    const float* wp = Wp + tid * 2 * D;
    for (int k = 0; k < 2 * D; ++k) acc += qstar[k] * wp[k];
    out[g * 3 + tid] = acc;
  }
}

// =====================================================================================
extern "C" void kernel_launch(void* const* d_in, const int* in_sizes, int n_in,
                              void* d_out, int out_size, void* d_ws, size_t ws_size,
                              hipStream_t stream) {
  const float* feats  = (const float*)d_in[0];
  const float* W_edge = (const float*)d_in[1];
  const float* b_edge = (const float*)d_in[2];
  const float* W_ih   = (const float*)d_in[3];
  const float* W_hh   = (const float*)d_in[4];
  const float* b_ih   = (const float*)d_in[5];
  const float* b_hh   = (const float*)d_in[6];
  const float* Wi0 = (const float*)d_in[7],  *Wh0 = (const float*)d_in[8];
  const float* bi0 = (const float*)d_in[9],  *bh0 = (const float*)d_in[10];
  const float* Wi1 = (const float*)d_in[11], *Wh1 = (const float*)d_in[12];
  const float* bi1 = (const float*)d_in[13], *bh1 = (const float*)d_in[14];
  const float* Wi2 = (const float*)d_in[15], *Wh2 = (const float*)d_in[16];
  const float* bi2 = (const float*)d_in[17], *bh2 = (const float*)d_in[18];
  const float* Wp  = (const float*)d_in[19], *bp  = (const float*)d_in[20];
  const int* src    = (const int*)d_in[21];
  const int* dstp   = (const int*)d_in[22];
  const int* etypes = (const int*)d_in[23];

  const int N = in_sizes[0] / D;          // 64000
  const int G = N / NPG;                  // 320
  const int E = in_sizes[21];             // 1,024,000
  const int epg = E / G;                  // 3200 (edges contiguous per graph)

  float* feat_mid = (float*)d_ws;         // [N][64] f32

  const size_t dyn_lds = (size_t)(2 * ROWS * D * 4 + ROWS * 192 * 4); // 266,240 B
  (void)hipFuncSetAttribute((const void*)ggnn_propagate,
                            hipFuncAttributeMaxDynamicSharedMemorySize, (int)dyn_lds);

  ggnn_propagate<<<G, 256, dyn_lds, stream>>>(feats, W_edge, b_edge, W_ih, W_hh, b_ih, b_hh,
                                              src, dstp, etypes, epg, feat_mid);
  set2set_readout<<<G, 256, 0, stream>>>(feat_mid,
                                         Wi0, Wh0, bi0, bh0,
                                         Wi1, Wh1, bi1, bh1,
                                         Wi2, Wh2, bi2, bh2,
                                         Wp, bp, (float*)d_out);
}
